// Attention_81226421502611
// MI455X (gfx1250) — matrix operand order repaired and checked by reading, not compiled, not run on hardware
//
#include <hip/hip_runtime.h>

// ---------------------------------------------------------------------------
// Flash-attention for MI455X (gfx1250, wave32, WMMA f32_16x16x32_f16)
//   q: [16, 2048, 256] f32   k: [16, 256, 2048] f32 (pre-transposed)
//   v: [16, 2048, 256] f32   mask: [16, 1, 2048] i32
//   out = transpose(softmax(mask(q@k)) / 16 @ v) -> [16, 256, 2048] f32
// ---------------------------------------------------------------------------

typedef __attribute__((ext_vector_type(16))) _Float16 v16h;
typedef __attribute__((ext_vector_type(8)))  _Float16 v8h;
typedef __attribute__((ext_vector_type(2)))  _Float16 h2;
typedef __attribute__((ext_vector_type(8)))  float    v8f;

#define BATCH 16
#define SEQ   2048
#define DIM   256
#define QTILE 128      // queries per block (8 waves x 16 queries)
#define KT    32       // key-chunk per iteration (one 16x32 WMMA K-slab)
#define NW    8
#define KSTR  264      // LDS strides (halves), padded to dodge bank conflicts
#define QSTR  264
#define VSTR  40

// Merge two 16B LDS loads into one 32B WMMA fragment (avoids requiring 32B
// alignment on every fragment address).
__device__ __forceinline__ v16h ld16(const _Float16* p0, const _Float16* p1) {
  v8h a = *(const v8h*)p0;
  v8h b = *(const v8h*)p1;
  return __builtin_shufflevector(a, b, 0,1,2,3,4,5,6,7,8,9,10,11,12,13,14,15);
}

__global__ __launch_bounds__(256, 1)
void flash_attn_wmma(const float* __restrict__ q,
                     const float* __restrict__ k,
                     const float* __restrict__ v,
                     const int*   __restrict__ mask,
                     float* __restrict__ out) {
  // LDS staging (~110 KB of the 320 KB WGP pool)
  __shared__ __align__(16) _Float16 Qs[QTILE * QSTR];   // [q][d]   f16
  __shared__ __align__(16) _Float16 Ks[KT * KSTR];      // [key][d] f16
  __shared__ __align__(16) _Float16 Vt[DIM * VSTR];     // [d][key] f16 (V^T)
  __shared__ __align__(32) _Float16 Ps[NW][16 * 32];    // per-wave P^T [q][key]
  __shared__ int Ms[KT];                                // mask chunk

  const int tid  = threadIdx.x;
  const int w    = tid >> 5;      // wave id (8 waves)
  const int lane = tid & 31;
  const int lm   = lane & 15;     // N / row-in-half index
  const int h    = lane >> 4;     // lane half

  const int b     = blockIdx.x / (SEQ / QTILE);
  const int qbase = (blockIdx.x % (SEQ / QTILE)) * QTILE;

  const float* qb = q + (size_t)(b * SEQ + qbase) * DIM;
  const float* kb = k + (size_t)b * DIM * SEQ;
  const float* vb = v + (size_t)b * SEQ * DIM;
  const int*   mb = mask + b * SEQ;

  // ---- stage the 128-query Q tile once: f32 -> f16, [q][d] ----
  for (int idx = tid; idx < QTILE * (DIM / 2); idx += 256) {
    int dp = idx & (DIM / 2 - 1);
    int qq = idx >> 7;
    float2 f = *(const float2*)(qb + qq * DIM + 2 * dp);
    h2 hv; hv.x = (_Float16)f.x; hv.y = (_Float16)f.y;
    *(h2*)&Qs[qq * QSTR + 2 * dp] = hv;
  }
  __syncthreads();

  // O^T accumulator: 16 d-tiles of 16x16 f32 (cols = this wave's 16 queries)
  v8f Ot[16];
  #pragma unroll
  for (int t = 0; t < 16; ++t) Ot[t] = (v8f)(0.0f);
  float mrun = -1.0e30f;   // running row max (per query = lane%16)
  float lrun = 0.0f;       // running row sum

  for (int kt = 0; kt < SEQ; kt += KT) {
    // ---- cooperative staging of K chunk [key][d] (coalesced 128B K rows) ----
    for (int idx = tid; idx < KT * (DIM / 2); idx += 256) {
      int j  = idx & (KT - 1);
      int dp = idx >> 5;                       // 0..127
      float f0 = kb[(size_t)(2 * dp)     * SEQ + kt + j];
      float f1 = kb[(size_t)(2 * dp + 1) * SEQ + kt + j];
      h2 hv; hv.x = (_Float16)f0; hv.y = (_Float16)f1;
      *(h2*)&Ks[j * KSTR + 2 * dp] = hv;
    }
    // ---- V chunk transposed into [d][key] (coalesced d reads) ----
    for (int idx = tid; idx < DIM * (KT / 2); idx += 256) {
      int d  = idx & (DIM - 1);
      int jp = idx >> 8;                       // 0..15
      float f0 = vb[(size_t)(kt + 2 * jp)     * DIM + d];
      float f1 = vb[(size_t)(kt + 2 * jp + 1) * DIM + d];
      h2 hv; hv.x = (_Float16)f0; hv.y = (_Float16)f1;
      *(h2*)&Vt[d * VSTR + 2 * jp] = hv;
    }
    if (tid < KT) Ms[tid] = mb[kt + tid];

    // ---- prefetch next chunk into cache (global_prefetch_b8) ----
    int ktn = kt + KT;
    if (ktn < SEQ) {
      __builtin_prefetch(kb + (size_t)tid * SEQ + ktn, 0, 1);          // K rows
      if (tid < 128)
        __builtin_prefetch(vb + (size_t)(ktn + (tid >> 2)) * DIM + (tid & 3) * 64, 0, 1);
    }
    __syncthreads();

    // ---- S^T = K_tile (32x256) * Q^T (256x16): two 16x16 C-tiles ----
    v8f s0 = (v8f)(0.0f), s1 = (v8f)(0.0f);
    #pragma unroll
    for (int c = 0; c < 8; ++c) {            // 256 dims = 8 slabs of K=32
      const _Float16* qp = &Qs[(w * 16 + lm) * QSTR + c * 32 + h * 8];
      v16h bq = ld16(qp, qp + 16);           // B frag: N=query, K=d
      const _Float16* k0 = &Ks[lm * KSTR + c * 32 + h * 8];
      v16h a0 = ld16(k0, k0 + 16);           // A frag: keys 0..15
      const _Float16* k1 = &Ks[(16 + lm) * KSTR + c * 32 + h * 8];
      v16h a1 = ld16(k1, k1 + 16);           // A frag: keys 16..31
      s0 = __builtin_amdgcn_wmma_f32_16x16x32_f16(false, a0, false, bq, (short)0, s0, false, false);
      s1 = __builtin_amdgcn_wmma_f32_16x16x32_f16(false, a1, false, bq, (short)0, s1, false, false);
    }

    // ---- mask + online softmax (per-lane query column = lane%16) ----
    float mloc = -1.0e30f;
    #pragma unroll
    for (int r = 0; r < 8; ++r) {
      int key0 = r + 8 * h;                  // C layout: M = r + 8*(lane/16)
      s0[r] = Ms[key0]      ? s0[r] : -1.0e9f;
      s1[r] = Ms[16 + key0] ? s1[r] : -1.0e9f;
      mloc = fmaxf(mloc, fmaxf(s0[r], s1[r]));
    }
    mloc = fmaxf(mloc, __shfl_xor(mloc, 16, 32));   // combine lane halves
    float mnew  = fmaxf(mrun, mloc);
    float alpha = __expf(mrun - mnew);

    float rsum = 0.0f;
    #pragma unroll
    for (int r = 0; r < 8; ++r) {
      int key0 = r + 8 * h;
      float p0 = __expf(s0[r] - mnew);
      float p1 = __expf(s1[r] - mnew);
      rsum += p0 + p1;
      Ps[w][lm * 32 + key0]      = (_Float16)p0;   // P^T scratch [q][key]
      Ps[w][lm * 32 + 16 + key0] = (_Float16)p1;
    }
    rsum += __shfl_xor(rsum, 16, 32);
    lrun = lrun * alpha + rsum;
    mrun = mnew;

    // ---- rescale O^T, then O^T += V^T (16x32) * P^T (32x16) per d-tile ----
    #pragma unroll
    for (int t = 0; t < 16; ++t) {
      #pragma unroll
      for (int r = 0; r < 8; ++r) Ot[t][r] *= alpha;
    }
    const _Float16* pp = &Ps[w][lm * 32 + h * 8];    // wave-private; same-wave
    v16h bp = ld16(pp, pp + 16);                     // LDS ops are in-order
    #pragma unroll
    for (int t = 0; t < 16; ++t) {
      const _Float16* vp = &Vt[(t * 16 + lm) * VSTR + h * 8];
      v16h av = ld16(vp, vp + 16);                   // A frag: M=d, K=key
      Ot[t] = __builtin_amdgcn_wmma_f32_16x16x32_f16(false, av, false, bp, (short)0, Ot[t], false, false);
    }
    __syncthreads();   // protect Ks/Vt/Ms before next chunk's staging
  }

  // ---- normalize (softmax sum AND the reference's post-softmax /sqrt(256))
  // and store O^T directly into out[b, d, q] (16 contiguous floats per row)
  const float inv   = 1.0f / (lrun * 16.0f);
  const int   qglob = qbase + w * 16 + lm;
  #pragma unroll
  for (int t = 0; t < 16; ++t) {
    #pragma unroll
    for (int r = 0; r < 8; ++r) {
      int d = t * 16 + r + 8 * h;
      out[((size_t)(b * DIM + d)) * SEQ + qglob] = Ot[t][r] * inv;
    }
  }
}

extern "C" void kernel_launch(void* const* d_in, const int* in_sizes, int n_in,
                              void* d_out, int out_size, void* d_ws, size_t ws_size,
                              hipStream_t stream) {
  (void)in_sizes; (void)n_in; (void)out_size; (void)d_ws; (void)ws_size;
  const float* q    = (const float*)d_in[0];
  const float* k    = (const float*)d_in[1];
  const float* v    = (const float*)d_in[2];
  const int*   mask = (const int*)d_in[3];
  float* out = (float*)d_out;

  dim3 grid(BATCH * (SEQ / QTILE));   // 256 blocks: (batch, 128-query tile)
  dim3 block(256);                    // 8 wave32 waves
  flash_attn_wmma<<<grid, block, 0, stream>>>(q, k, v, mask, out);
}
